// SpectralAwareSSM_13941463843633
// MI455X (gfx1250) — matrix operand order repaired and checked by reading, not compiled, not run on hardware
//
#include <hip/hip_runtime.h>
#include <hip/hip_bf16.h>

// ---------------------------------------------------------------------------
// SpectralAwareSSM on MI455X (gfx1250), fully fused 4-kernel pipeline:
//   1) proj_kernel : WMMA f16 GEMM x@x_proj_w^T (33 outs) + fused snr proj /
//                    sigmoid gating  -> 33-float record per (b,l) timestep.
//                    Weights staged via async global->LDS (ASYNCcnt).
//   2) scan_pass<false> : chunked local scan (16 chunks of 64 steps),
//                    emits per-chunk decay product P and local end state.
//   3) chunk_carry : sequential 16-step carry across chunks (4096 threads).
//   4) scan_pass<true>  : re-run local scan seeded with correct h0, emit y.
// Never materializes dA/dBx (would be 268 MB each); total HBM traffic ~65 MB.
// ---------------------------------------------------------------------------

typedef __attribute__((ext_vector_type(16))) _Float16 v16h;
typedef __attribute__((ext_vector_type(8)))  float    v8f;

#define D_INNER 512
#define D_STATE 16
#define N_MELS  40
#define BATCH   8
#define SEQ     1024
#define ROWS    (BATCH * SEQ)     // 8192 timesteps
#define NPROJ   33                // dt_raw(1) + B(16) + C(16)
#define REC_STRIDE 36             // padded record stride (floats)
#define NC      16                // L chunks
#define CHUNK   (SEQ / NC)        // 64

// LDS layout for proj_kernel (dynamic shared):
//   [0, 49152)           : f16 weight image wh[48][512] (rows 33..47 unused)
//   [49152, 49152+67584) : union { f32 staging of x_proj_w (pre-loop)
//                                  f32 tile scratch scr[128][48] (post-loop) }
//   [116736, ...)        : snr weights (17*40) + bias (17), f32
#define WH_BYTES   (48 * 512 * 2)
#define STAGE_OFF  WH_BYTES
#define STAGE_BYTES (NPROJ * 512 * 4)
#define SNR_OFF    (STAGE_OFF + STAGE_BYTES)
#define SNRW_N     (17 * N_MELS)
#define SMEM_BYTES (SNR_OFF + (SNRW_N + 17) * 4)   // ~119.5 KB of 320 KB/WGP

__global__ void __launch_bounds__(256) proj_kernel(
    const float* __restrict__ x, const float* __restrict__ snr_mel,
    const float* __restrict__ xw, const float* __restrict__ sw,
    const float* __restrict__ sb, const float* __restrict__ alpha_p,
    const float* __restrict__ gf_p, float* __restrict__ rec)
{
  extern __shared__ char smem[];
  _Float16* wh    = (_Float16*)smem;
  float*    stage = (float*)(smem + STAGE_OFF);
  float*    scr   = (float*)(smem + STAGE_OFF);   // reused after staging
  float*    snrW  = (float*)(smem + SNR_OFF);
  float*    snrB  = snrW + SNRW_N;
  const int tid = threadIdx.x;

  // ---- async global->LDS staging of x_proj_w (f32), CDNA5 ASYNCcnt path ----
  for (int i = tid; i < NPROJ * 512; i += 256) {
    unsigned lds = (unsigned)(unsigned long long)(&stage[i]);  // low 32 bits = LDS offset
    unsigned long long ga = (unsigned long long)(xw + i);
    asm volatile("global_load_async_to_lds_b32 %0, %1, off"
                 :: "v"(lds), "v"(ga) : "memory");
  }
  for (int i = tid; i < SNRW_N; i += 256) snrW[i] = sw[i];
  if (tid < 17) snrB[tid] = sb[tid];
  asm volatile("s_wait_asynccnt 0x0" ::: "memory");
  __syncthreads();

  // one-time f32 -> f16 conversion into the B-fragment-friendly layout
  // wh[n][k] (n = feature row of x_proj_w, k contiguous)
  for (int i = tid; i < NPROJ * 512; i += 256) wh[i] = (_Float16)stage[i];
  __syncthreads();

  // ---- WMMA GEMM: each wave owns a 16-row x 48-col tile, K=512 ----
  const int wave = tid >> 5, lane = tid & 31;
  const int g = lane >> 4;        // lane group (K-half select)
  const int m = lane & 15;        // A row / B column within tile
  const int rowBase = blockIdx.x * 128 + wave * 16;
  const float* xrow = x + (size_t)(rowBase + m) * D_INNER;

  v8f acc0 = {}, acc1 = {}, acc2 = {};
  for (int kb = 0; kb < 512; kb += 32) {
    // A fragment (contiguous k order, same as B): lane holds K kb+16g..kb+16g+15
    const int k0 = kb + 16 * g;
    float4 f0 = *(const float4*)(xrow + k0);
    float4 f1 = *(const float4*)(xrow + k0 + 4);
    float4 f2 = *(const float4*)(xrow + k0 + 8);
    float4 f3 = *(const float4*)(xrow + k0 + 12);
    v16h a;
    a[0]=(_Float16)f0.x;  a[1]=(_Float16)f0.y;  a[2]=(_Float16)f0.z;  a[3]=(_Float16)f0.w;
    a[4]=(_Float16)f1.x;  a[5]=(_Float16)f1.y;  a[6]=(_Float16)f1.z;  a[7]=(_Float16)f1.w;
    a[8]=(_Float16)f2.x;  a[9]=(_Float16)f2.y;  a[10]=(_Float16)f2.z; a[11]=(_Float16)f2.w;
    a[12]=(_Float16)f3.x; a[13]=(_Float16)f3.y; a[14]=(_Float16)f3.z; a[15]=(_Float16)f3.w;

    // B fragments: lane group g holds K = kb+16g..kb+16g+15 of column n=m.
    // wh row-major [feature][k] makes this one contiguous 32-byte LDS load.
    const int bo = kb + 16 * g;
    v16h b0 = *(const v16h*)(wh + (     m) * 512 + bo);
    v16h b1 = *(const v16h*)(wh + (16 + m) * 512 + bo);
    v16h b2 = *(const v16h*)(wh + (32 + m) * 512 + bo);   // cols >=33 discarded

    acc0 = __builtin_amdgcn_wmma_f32_16x16x32_f16(false, a, false, b0, (short)0, acc0, false, false);
    acc1 = __builtin_amdgcn_wmma_f32_16x16x32_f16(false, a, false, b1, (short)0, acc1, false, false);
    acc2 = __builtin_amdgcn_wmma_f32_16x16x32_f16(false, a, false, b2, (short)0, acc2, false, false);
  }

  // scatter D tiles to LDS so each timestep's 33 features land in one row
  #pragma unroll
  for (int v = 0; v < 8; ++v) {
    const int r = wave * 16 + v + 8 * g;   // C/D layout: M = v + 8*(lane>=16)
    scr[r * 48 +      m] = acc0[v];
    scr[r * 48 + 16 + m] = acc1[v];
    scr[r * 48 + 32 + m] = acc2[v];
  }
  __syncthreads();

  // ---- fused SNR projection + gating epilogue: one thread per timestep ----
  if (tid < 128) {
    const int grow = blockIdx.x * 128 + tid;
    const float* mel = snr_mel + (size_t)grow * N_MELS;
    float melv[N_MELS];
    #pragma unroll
    for (int i = 0; i < N_MELS / 4; ++i) {
      float4 t = *(const float4*)(mel + 4 * i);
      melv[4*i] = t.x; melv[4*i+1] = t.y; melv[4*i+2] = t.z; melv[4*i+3] = t.w;
    }
    float smod[17];
    #pragma unroll
    for (int p = 0; p < 17; ++p) {
      float acc = snrB[p];
      const float* wp = snrW + p * N_MELS;
      #pragma unroll
      for (int j = 0; j < N_MELS; ++j) acc += wp[j] * melv[j];
      smod[p] = acc;
    }
    const float alpha = alpha_p[0], gf = gf_p[0];
    const float* row = scr + tid * 48;
    float* out = rec + (size_t)grow * REC_STRIDE;
    out[0] = row[0] + smod[0];                         // s = dt_raw + dt_snr_shift
    #pragma unroll
    for (int j = 0; j < D_STATE; ++j) {
      const float raw  = 1.0f / (1.0f + __expf(-smod[1 + j]));
      const float gate = gf + (1.0f - gf) * raw;
      out[1 + j]  = row[1 + j] * (1.0f - alpha + alpha * gate);  // gated B
      out[17 + j] = row[17 + j];                                  // C
    }
  }
}

// ---------------------------------------------------------------------------
// Chunked scan. FINAL=false: local scan from h=0, emit P=prod(dA) and h_end.
// FINAL=true : local scan from Hinit (correct carry), emit y.
// grid = (D/64, B, NC), block = 64 (one channel per lane -> 2048 waves total).
// ---------------------------------------------------------------------------
template <bool FINAL>
__global__ void __launch_bounds__(64) scan_pass(
    const float* __restrict__ x,    const float* __restrict__ rec,
    const float* __restrict__ dtw,  const float* __restrict__ dtb,
    const float* __restrict__ A_log,const float* __restrict__ Dp,
    const float* __restrict__ Hinit, float* __restrict__ Pst,
    float* __restrict__ hend, float* __restrict__ y)
{
  __shared__ float srec[CHUNK * REC_STRIDE];   // 9 KB: this chunk's records
  const int tid = threadIdx.x;
  const int d = blockIdx.x * 64 + tid;
  const int b = blockIdx.y;
  const int c = blockIdx.z;

  const size_t recBase = ((size_t)b * SEQ + (size_t)c * CHUNK) * REC_STRIDE;
  for (int i = tid; i < CHUNK * REC_STRIDE; i += 64) srec[i] = rec[recBase + i];
  __syncthreads();

  float Ar[D_STATE], h[D_STATE], P[D_STATE];
  const size_t hBase = ((((size_t)c * BATCH + b) * D_INNER) + d) * D_STATE;
  #pragma unroll
  for (int n = 0; n < D_STATE; ++n) {
    Ar[n] = -__expf(A_log[d * D_STATE + n]);   // A = -exp(A_log)
    h[n]  = FINAL ? Hinit[hBase + n] : 0.0f;
    P[n]  = 1.0f;
  }
  const float wd = dtw[d], bd = dtb[d];
  const float dval = FINAL ? Dp[d] : 0.0f;
  const size_t xBase = ((size_t)b * SEQ + (size_t)c * CHUNK) * D_INNER + d;

  for (int l = 0; l < CHUNK; ++l) {
    const float xv = x[xBase + (size_t)l * D_INNER];      // coalesced over d
    const float* r = srec + l * REC_STRIDE;
    const float z = r[0] * wd + bd;
    const float delta = (z > 20.0f) ? z : log1pf(__expf(z));  // softplus
    const float dx = delta * xv;
    float acc = dval * xv;
    #pragma unroll
    for (int n = 0; n < D_STATE; ++n) {
      const float dA = __expf(Ar[n] * delta);
      h[n] = dA * h[n] + dx * r[1 + n];
      if (FINAL) acc += h[n] * r[17 + n];
      else       P[n] *= dA;
    }
    if (FINAL) y[xBase + (size_t)l * D_INNER] = acc;
  }
  if (!FINAL) {
    #pragma unroll
    for (int n = 0; n < D_STATE; ++n) {
      Pst[hBase + n]  = P[n];
      hend[hBase + n] = h[n];
    }
  }
}

// Sequential carry across the 16 chunks: H_{c+1} = P_c * H_c + hend_c.
__global__ void __launch_bounds__(256) chunk_carry(
    const float* __restrict__ Pst, const float* __restrict__ hend,
    float* __restrict__ Hinit)
{
  const int id = blockIdx.x * 256 + threadIdx.x;  // 0..4095
  const int b = id >> 9, d = id & 511;
  float H[D_STATE];
  #pragma unroll
  for (int n = 0; n < D_STATE; ++n) H[n] = 0.0f;
  for (int c = 0; c < NC; ++c) {
    const size_t base = ((((size_t)c * BATCH + b) * D_INNER) + d) * D_STATE;
    #pragma unroll
    for (int n = 0; n < D_STATE; ++n) {
      Hinit[base + n] = H[n];                       // h0 for chunk c
      H[n] = Pst[base + n] * H[n] + hend[base + n]; // advance to chunk c+1
    }
  }
}

extern "C" void kernel_launch(void* const* d_in, const int* in_sizes, int n_in,
                              void* d_out, int out_size, void* d_ws, size_t ws_size,
                              hipStream_t stream) {
  const float* x     = (const float*)d_in[0];
  const float* mel   = (const float*)d_in[1];
  const float* xw    = (const float*)d_in[2];
  const float* sw    = (const float*)d_in[3];
  const float* sb    = (const float*)d_in[4];
  const float* dtw   = (const float*)d_in[5];
  const float* dtb   = (const float*)d_in[6];
  const float* Alog  = (const float*)d_in[7];
  const float* Dp    = (const float*)d_in[8];
  const float* alpha = (const float*)d_in[9];
  const float* gf    = (const float*)d_in[10];
  float* y = (float*)d_out;

  // workspace: records (1.2 MB) + P/hend/Hinit (4.2 MB each) = ~13.8 MB
  float* ws    = (float*)d_ws;
  float* rec   = ws;
  float* Pst   = rec  + (size_t)ROWS * REC_STRIDE;
  float* hend  = Pst  + (size_t)NC * BATCH * D_INNER * D_STATE;
  float* Hinit = hend + (size_t)NC * BATCH * D_INNER * D_STATE;

  proj_kernel<<<ROWS / 128, 256, SMEM_BYTES, stream>>>(x, mel, xw, sw, sb, alpha, gf, rec);

  dim3 sg(D_INNER / 64, BATCH, NC);
  scan_pass<false><<<sg, 64, 0, stream>>>(x, rec, dtw, dtb, Alog, Dp,
                                          nullptr, Pst, hend, nullptr);
  chunk_carry<<<(BATCH * D_INNER) / 256, 256, 0, stream>>>(Pst, hend, Hinit);
  scan_pass<true><<<sg, 64, 0, stream>>>(x, rec, dtw, dtb, Alog, Dp,
                                         Hinit, nullptr, nullptr, y);
}
